// MonotonicAttentionLayer_17952963297959
// MI455X (gfx1250) — matrix operand order repaired and checked by reading, not compiled, not run on hardware
//
#include <hip/hip_runtime.h>

typedef __attribute__((ext_vector_type(16))) __bf16 v16bf;
typedef __attribute__((ext_vector_type(8)))  __bf16 v8bf;
typedef __attribute__((ext_vector_type(8)))  float  v8f;

#define S_DIM 2048
#define B_DIM 32
#define E_DIM 1024
#define A_DIM 1024
#define SX_STRIDE 1040   // 1024 bf16 + 16 pad (32B) to spread LDS banks

__device__ __forceinline__ unsigned short f2bf(float f) {
    unsigned int u = __float_as_uint(f);
    u += 0x7FFFu + ((u >> 16) & 1u);   // round-to-nearest-even
    return (unsigned short)(u >> 16);
}

// Hardware tanh (CDNA5 TRANS op) — confirmed last round (valu 4287 -> 761).
__device__ __forceinline__ float fast_tanh(float x) {
#if __has_builtin(__builtin_amdgcn_tanhf)
    return __builtin_amdgcn_tanhf(x);
#else
    float y;
    asm volatile("v_tanh_f32 %0, %1" : "=v"(y) : "v"(x));
    return y;
#endif
}

// ---------------------------------------------------------------------------
// k0: repack W_enc [E][A] f32 -> bf16 WMMA B-fragment layout:
//     Wfrag[(ntile*32 + kstep)*32 + lane][16 bf16]
//     lane: N = ntile*16 + (lane&15); elem j -> K = kstep*32 + (lane>>4)*16 + j
// ---------------------------------------------------------------------------
__global__ void k_pack_wenc(const float* __restrict__ W_enc,
                            unsigned short* __restrict__ Wfrag) {
    int idx   = blockIdx.x * blockDim.x + threadIdx.x;   // 0..65535
    int lane  = idx & 31;
    int kstep = (idx >> 5) & 31;
    int ntile = idx >> 10;
    int laneN = lane & 15, laneG = lane >> 4;
    const float* src = W_enc + (ntile * 16 + laneN);
    unsigned int pk[8];
#pragma unroll
    for (int jj = 0; jj < 8; ++jj) {
        int k0 = kstep * 32 + laneG * 16 + jj * 2;
        unsigned short lo = f2bf(src[(size_t)k0 * A_DIM]);
        unsigned short hi = f2bf(src[(size_t)(k0 + 1) * A_DIM]);
        pk[jj] = (unsigned int)lo | ((unsigned int)hi << 16);
    }
    uint4* dst = (uint4*)(Wfrag + (size_t)idx * 16);
    dst[0] = make_uint4(pk[0], pk[1], pk[2], pk[3]);
    dst[1] = make_uint4(pk[4], pk[5], pk[6], pk[7]);
}

// ---------------------------------------------------------------------------
// k1: bias2[b][a] = decoder_state[b,:] @ W_dec[:,a] + b[a]   (tiny GEMM)
//     bb derived from blockIdx only -> decoder row goes through scalar loads.
// ---------------------------------------------------------------------------
__global__ void k_dec_proj(const float* __restrict__ dec,
                           const float* __restrict__ W_dec,
                           const float* __restrict__ bvec,
                           float* __restrict__ bias2) {
    const int bb = blockIdx.x >> 2;                        // uniform per block
    const int a  = (blockIdx.x & 3) * 256 + threadIdx.x;   // 0..1023
    float acc = bvec[a];
    const float* dr = dec + (size_t)bb * 1024;
    const float* wd = W_dec + a;
#pragma unroll 4
    for (int d = 0; d < 1024; ++d)
        acc = fmaf(dr[d], wd[(size_t)d * A_DIM], acc);
    bias2[bb * A_DIM + a] = acc;
}

// ---------------------------------------------------------------------------
// k2: main fused kernel. Per block: 32 rows of X=[S*B, E].
//     energy[r] = sum_a tanh((X @ W_enc)[r,a] + bias2[b(r),a]) * v[a]
//     p[r]      = sigmoid(energy[r] + noise[r])
// ---------------------------------------------------------------------------
__global__ void __launch_bounds__(256)
k_energy(const float* __restrict__ enc, const float* __restrict__ noise,
         const unsigned short* __restrict__ Wfrag,
         const float* __restrict__ bias2, const float* __restrict__ v,
         float* __restrict__ P) {
    extern __shared__ unsigned char smem[];
    unsigned short* sX = (unsigned short*)smem;               // [32][SX_STRIDE] bf16
    float* sEnergy = (float*)(smem + 32 * SX_STRIDE * 2);     // [32]

    const int tid = threadIdx.x;
    const int rowBase = blockIdx.x * 32;                      // multiple of 32 -> b == local row

    if (tid < 32) sEnergy[tid] = 0.0f;

    // Stage 32x1024 f32 tile -> bf16 LDS
    for (int i = tid; i < 32 * 256; i += 256) {
        int row = i >> 8;
        int c4  = i & 255;
        const float4 f = *(const float4*)(enc + ((size_t)(rowBase + row) * E_DIM + c4 * 4));
        uint2 pk;
        pk.x = (unsigned int)f2bf(f.x) | ((unsigned int)f2bf(f.y) << 16);
        pk.y = (unsigned int)f2bf(f.z) | ((unsigned int)f2bf(f.w) << 16);
        *(uint2*)(sX + row * SX_STRIDE + c4 * 4) = pk;
    }
    __syncthreads();

    const int wave = tid >> 5;
    const int lane = tid & 31;
    const int laneN = lane & 15;
    const int laneG = lane >> 4;
    const int ntBase = wave * 8;        // 8 waves x 8 N-tiles = 64 tiles of A

    v8f acc0[8], acc1[8];
#pragma unroll
    for (int nt = 0; nt < 8; ++nt) { acc0[nt] = {}; acc1[nt] = {}; }

    const unsigned short* srowA0 = sX + laneN * SX_STRIDE;          // rows 0-15
    const unsigned short* srowA1 = sX + (laneN + 16) * SX_STRIDE;   // rows 16-31

    // Single per-lane base; every (nt,k) fragment is a CONSTANT byte offset
    // (nt*32+k)*1024 < 2^18, well inside the signed 24-bit IOFFSET field, so
    // the compiler can encode them as global_load immediates.
    const unsigned short* wbase = Wfrag + ((size_t)(ntBase * 32) * 32 + lane) * 16;

    for (int k = 0; k < 32; ++k) {
        // A-fragment (matches Wfrag contiguous layout): lane group kg -> K runs
        // [kg*16 .. kg*16+15] : two aligned 16B LDS loads
        const int c0 = k * 32 + laneG * 16;
        v8bf a0lo = *(const v8bf*)(srowA0 + c0);
        v8bf a0hi = *(const v8bf*)(srowA0 + c0 + 8);
        v8bf a1lo = *(const v8bf*)(srowA1 + c0);
        v8bf a1hi = *(const v8bf*)(srowA1 + c0 + 8);
        v16bf aF0 = __builtin_shufflevector(a0lo, a0hi, 0,1,2,3,4,5,6,7,8,9,10,11,12,13,14,15);
        v16bf aF1 = __builtin_shufflevector(a1lo, a1hi, 0,1,2,3,4,5,6,7,8,9,10,11,12,13,14,15);
#pragma unroll
        for (int nt = 0; nt < 8; ++nt) {
            const v16bf bF = *(const v16bf*)(wbase + (nt * 32 + k) * 512);
            acc0[nt] = __builtin_amdgcn_wmma_f32_16x16x32_bf16(
                false, aF0, false, bF, (short)0, acc0[nt], false, false);
            acc1[nt] = __builtin_amdgcn_wmma_f32_16x16x32_bf16(
                false, aF1, false, bF, (short)0, acc1[nt], false, false);
        }
    }

    // Epilogue: bias + tanh + dot with v, per-lane accumulation over owned N-tiles
    float rsum[16];
#pragma unroll
    for (int i = 0; i < 16; ++i) rsum[i] = 0.0f;

#pragma unroll
    for (int nt = 0; nt < 8; ++nt) {
        const int aCol = (ntBase + nt) * 16 + laneN;
        const float vv = v[aCol];
#pragma unroll
        for (int i = 0; i < 8; ++i) {
            const int row0 = laneG * 8 + i;     // C layout: lanes 16-31 hold M=i+8
            const int row1 = row0 + 16;         // second row-tile
            float x0 = acc0[nt][i] + bias2[row0 * A_DIM + aCol];  // b == local row
            float x1 = acc1[nt][i] + bias2[row1 * A_DIM + aCol];
            rsum[i]     += fast_tanh(x0) * vv;
            rsum[i + 8] += fast_tanh(x1) * vv;
        }
    }

    // Reduce across the 16 N-lanes of each half-wave (xor masks stay in-half)
#pragma unroll
    for (int m = 1; m <= 8; m <<= 1) {
#pragma unroll
        for (int i = 0; i < 16; ++i) rsum[i] += __shfl_xor(rsum[i], m, 32);
    }
    if (laneN == 0) {
        const int base = laneG * 8;
#pragma unroll
        for (int i = 0; i < 8; ++i) {
            atomicAdd(&sEnergy[base + i],      rsum[i]);       // rows 0-15
            atomicAdd(&sEnergy[base + i + 16], rsum[i + 8]);   // rows 16-31
        }
    }
    __syncthreads();

    if (tid < 32) {
        const int r = rowBase + tid;
        const float e = sEnergy[tid] + noise[r];
        P[r] = 1.0f / (1.0f + __expf(-e));
    }
}

// ---------------------------------------------------------------------------
// k3: monotonic-attention closed form: alpha[s] = p[s] * prod_{j<s}(1-p[j]);
//     residual mass dumped on last frame. One lane per batch column.
// ---------------------------------------------------------------------------
__global__ void k_alpha(const float* __restrict__ P, float* __restrict__ alphaOut) {
    const int b = threadIdx.x;                 // 32 lanes
    float q = 1.0f, sum = 0.0f;
    for (int s = 0; s < S_DIM; ++s) {
        float p  = P[s * B_DIM + b];
        float al = p * q;
        alphaOut[s * B_DIM + b] = al;
        if (s < S_DIM - 1) sum += al;
        q *= (1.0f - p);
    }
    alphaOut[(S_DIM - 1) * B_DIM + b] = 1.0f - fminf(fmaxf(sum, 0.0f), 1.0f);
}

// ---------------------------------------------------------------------------
// k4/k5: context[b][e] = sum_s enc[s][b][e] * alpha[s][b], deterministic
//        two-stage reduction over 8 S-chunks (no float global atomics).
// ---------------------------------------------------------------------------
__global__ void k_ctx_partial(const float* __restrict__ enc,
                              const float* __restrict__ alpha,
                              float* __restrict__ partial) {
    const int e  = blockIdx.x * 256 + threadIdx.x;
    const int b  = blockIdx.y;
    const int sc = blockIdx.z;
    float acc = 0.0f;
    const int s0 = sc * 256;
#pragma unroll 4
    for (int s = s0; s < s0 + 256; ++s)
        acc = fmaf(enc[((size_t)(s * B_DIM + b)) * E_DIM + e], alpha[s * B_DIM + b], acc);
    partial[((size_t)sc * B_DIM + b) * E_DIM + e] = acc;
}

__global__ void k_ctx_reduce(const float* __restrict__ partial,
                             float* __restrict__ ctx) {
    const int idx = blockIdx.x * 256 + threadIdx.x;     // 0..32767
    float s = 0.0f;
#pragma unroll
    for (int sc = 0; sc < 8; ++sc) s += partial[(size_t)sc * (B_DIM * E_DIM) + idx];
    ctx[idx] = s;
}

// ---------------------------------------------------------------------------
extern "C" void kernel_launch(void* const* d_in, const int* in_sizes, int n_in,
                              void* d_out, int out_size, void* d_ws, size_t ws_size,
                              hipStream_t stream) {
    const float* dec   = (const float*)d_in[0];   // [32,1024]
    const float* enc   = (const float*)d_in[1];   // [2048,32,1024]
    const float* noise = (const float*)d_in[2];   // [2048,32]
    const float* W_dec = (const float*)d_in[3];   // [1024,1024]
    const float* W_enc = (const float*)d_in[4];   // [1024,1024]
    const float* bvec  = (const float*)d_in[5];   // [1024]
    const float* vvec  = (const float*)d_in[6];   // [1024]

    float* ctx      = (float*)d_out;                  // [32,1024]
    float* alphaOut = (float*)d_out + B_DIM * E_DIM;  // [2048,32]

    unsigned char* ws = (unsigned char*)d_ws;
    unsigned short* Wfrag = (unsigned short*)ws;                              // 2 MB
    float* bias2   = (float*)(ws + (2u << 20));                               // 128 KB
    float* P       = (float*)(ws + (2u << 20) + (128u << 10));                // 256 KB
    float* partial = (float*)(ws + (2u << 20) + (128u << 10) + (256u << 10)); // 1 MB

    k_pack_wenc<<<256, 256, 0, stream>>>(W_enc, Wfrag);
    k_dec_proj<<<128, 256, 0, stream>>>(dec, W_dec, bvec, bias2);

    const size_t smem = 32 * SX_STRIDE * 2 + 32 * sizeof(float);  // ~65.1 KB
    k_energy<<<2048, 256, smem, stream>>>(enc, noise, Wfrag, bias2, vvec, P);

    k_alpha<<<1, 32, 0, stream>>>(P, alphaOut);
    k_ctx_partial<<<dim3(4, 32, 8), 256, 0, stream>>>(enc, alphaOut, partial);
    k_ctx_reduce<<<128, 256, 0, stream>>>(partial, ctx);
}
